// GAT_30640296690075
// MI455X (gfx1250) — matrix operand order repaired and checked by reading, not compiled, not run on hardware
//
#include <hip/hip_runtime.h>
#include <hip/hip_bf16.h>

typedef __attribute__((ext_vector_type(16))) _Float16 v16h;
typedef __attribute__((ext_vector_type(8)))  _Float16 v8h;
typedef __attribute__((ext_vector_type(8)))  float    v8f;

#define B_  8
#define V_  2048
#define H_  256
#define D_  128
#define NEG_SLOPE 0.2f
#define NCHUNK (V_ / 32)

__device__ __forceinline__ float lrelu(float x) { return x > 0.0f ? x : NEG_SLOPE * x; }

// Convert a float4 into 4 consecutive f16 elements of a v16h (constant base index).
#define CVT4(DST, B0, S)                         \
    DST[(B0)+0] = (_Float16)(S).x;               \
    DST[(B0)+1] = (_Float16)(S).y;               \
    DST[(B0)+2] = (_Float16)(S).z;               \
    DST[(B0)+3] = (_Float16)(S).w;

// ---------------------------------------------------------------------------
// Kernel 0: one-time W (f32, DxH) -> Wh (f16) conversion.
// ---------------------------------------------------------------------------
__global__ __launch_bounds__(256) void gat_cvtw(const float* __restrict__ W,
                                                _Float16* __restrict__ Wh)
{
    const int i = blockIdx.x * 256 + threadIdx.x;   // D_*H_ = 32768 elements
    Wh[i] = (_Float16)W[i];
}

// ---------------------------------------------------------------------------
// Kernel 1: p = x @ W^T via WMMA f16->f32, stored transposed: pT[b][d][v].
// M-blocked x2: one wave computes a 32(v) x 128(d) tile -> each B-tile load
// feeds 2 WMMAs and A-rebuild hazards amortize over 16 WMMAs per ht step.
// 128 blocks x 128 threads (4 waves), 512 wave-tiles total.
// ---------------------------------------------------------------------------
__global__ __launch_bounds__(128) void gat_proj(const float* __restrict__ x,
                                                const _Float16* __restrict__ Wh,
                                                _Float16* __restrict__ pT)
{
    const int lane  = threadIdx.x & 31;
    const int wave  = threadIdx.x >> 5;
    const int tile2 = blockIdx.x * 4 + wave;    // 0..511, 32 rows each
    const int b     = tile2 >> 6;               // / (V/32)
    const int v0    = (tile2 & 63) << 5;
    const int row   = lane & 15;
    const int hi    = lane >> 4;

    const v8f zero = {};
    v8f acc[2][8];
#pragma unroll
    for (int k = 0; k < 2; ++k)
#pragma unroll
        for (int i = 0; i < 8; ++i) acc[k][i] = zero;

    const float* xr0 = x + ((size_t)b * V_ + (v0 + row)) * H_;
    const float* xr1 = xr0 + 16 * H_;

    for (int ht = 0; ht < H_ / 32; ++ht) {
        // A tiles (16x32): elems 0..15 -> h = hb+16*hi+0..15 (contiguous)
        v16h A[2];
#pragma unroll
        for (int k = 0; k < 2; ++k) {
            const float* ap = (k ? xr1 : xr0) + ht * 32 + 16 * hi;
            float4 a0 = *(const float4*)(ap + 0);
            float4 a1 = *(const float4*)(ap + 4);
            float4 a2 = *(const float4*)(ap + 8);
            float4 a3 = *(const float4*)(ap + 12);
            CVT4(A[k], 0, a0) CVT4(A[k], 4, a1) CVT4(A[k], 8, a2) CVT4(A[k], 12, a3)
        }

#pragma unroll
        for (int dt = 0; dt < 8; ++dt) {
            // B tile (32x16): lane = column d, K = 16*hi + 0..15 contiguous f16
            const int d = dt * 16 + row;
            v16h Bm = *(const v16h*)(Wh + (size_t)d * H_ + ht * 32 + 16 * hi);
            acc[0][dt] = __builtin_amdgcn_wmma_f32_16x16x32_f16(
                false, A[0], false, Bm, (short)0, acc[0][dt], false, false);
            acc[1][dt] = __builtin_amdgcn_wmma_f32_16x16x32_f16(
                false, A[1], false, Bm, (short)0, acc[1][dt], false, false);
        }
    }

    // C layout: lane holds column d = dt*16+row, rows (v0+16k) + 8*hi + r.
    // Transposed store -> 8 consecutive f16 along v: one 16B store per (k,dt).
#pragma unroll
    for (int k = 0; k < 2; ++k)
#pragma unroll
        for (int dt = 0; dt < 8; ++dt) {
            const int d = dt * 16 + row;
            _Float16* dst = pT + ((size_t)b * D_ + d) * V_ + v0 + 16 * k + 8 * hi;
            v8h o;
#pragma unroll
            for (int r = 0; r < 8; ++r) o[r] = (_Float16)acc[k][dt][r];
            *(v8h*)dst = o;
        }
}

// ---------------------------------------------------------------------------
// Kernel 2: e[b][v] = p . a_left, f[b][v] = p . a_right  (coalesced over v).
// ---------------------------------------------------------------------------
__global__ __launch_bounds__(256) void gat_ef(const _Float16* __restrict__ pT,
                                              const float* __restrict__ a,
                                              float* __restrict__ e,
                                              float* __restrict__ f)
{
    const int idx = blockIdx.x * 256 + threadIdx.x;  // 0 .. B*V-1
    const int b   = idx >> 11;
    const int v   = idx & (V_ - 1);
    const _Float16* base = pT + (size_t)b * D_ * V_ + v;
    float accL = 0.0f, accR = 0.0f;
#pragma unroll 4
    for (int d = 0; d < D_; ++d) {
        float pv = (float)base[(size_t)d * V_];
        accL += pv * a[d];
        accR += pv * a[D_ + d];
    }
    e[idx] = accL;
    f[idx] = accR;
}

// ---------------------------------------------------------------------------
// Kernel 3: Mb[b] = max_v f[b][v]   (one block per batch)
// ---------------------------------------------------------------------------
__global__ __launch_bounds__(256) void gat_max(const float* __restrict__ f,
                                               float* __restrict__ Mb)
{
    __shared__ float red[256];
    const int b = blockIdx.x;
    float m = -3.4e38f;
    for (int v = threadIdx.x; v < V_; v += 256) m = fmaxf(m, f[b * V_ + v]);
    red[threadIdx.x] = m;
    __syncthreads();
    for (int s = 128; s > 0; s >>= 1) {
        if (threadIdx.x < s) red[threadIdx.x] = fmaxf(red[threadIdx.x], red[threadIdx.x + s]);
        __syncthreads();
    }
    if (threadIdx.x == 0) Mb[b] = red[0];
}

// ---------------------------------------------------------------------------
// Kernel 4: fused masked attention (flash-style, softmax denominator cancels):
//   out_i = relu( sum_j adj_ij * exp(lrelu(e_i+f_j) - m_i) * p_j / sum_j ... )
// M-blocked x2: one wave owns 32 rows (two 16-row tiles); j in chunks of 32.
// The 8KB pT panel per chunk is shared by all 4 waves of the workgroup:
// cooperatively DMA'd into LDS via async global->LDS loads, double-buffered.
// ---------------------------------------------------------------------------
__device__ __forceinline__ void async_panel_16B_x4(unsigned lds_addr, const void* gaddr)
{
    // INST_OFFSET applies to BOTH the LDS and global address (ISA async rules):
    // one LDS-addr VGPR + one 64-bit global addr covers 64B per lane.
    asm volatile(
        "global_load_async_to_lds_b128 %0, %1, off\n\t"
        "global_load_async_to_lds_b128 %0, %1, off offset:16\n\t"
        "global_load_async_to_lds_b128 %0, %1, off offset:32\n\t"
        "global_load_async_to_lds_b128 %0, %1, off offset:48"
        :: "v"(lds_addr), "v"(gaddr) : "memory");
}

// Build one 16x32 probability A-tile (masked exp weights), accumulating lsum.
__device__ __forceinline__ v16h build_A(float ei, float mi, float& lsum,
                                        float4 f0, float4 f1, float4 f2, float4 f3,
                                        int4 m0, int4 m1, int4 m2, int4 m3)
{
    v16h A;
#define WCOMP(IDX, MV, FV)                                              \
    { float wv_ = (MV) ? __expf(lrelu(ei + (FV)) - mi) : 0.0f;          \
      lsum += wv_; A[IDX] = (_Float16)wv_; }
    WCOMP(0,  m0.x, f0.x) WCOMP(1,  m0.y, f0.y) WCOMP(2,  m0.z, f0.z) WCOMP(3,  m0.w, f0.w)
    WCOMP(4,  m1.x, f1.x) WCOMP(5,  m1.y, f1.y) WCOMP(6,  m1.z, f1.z) WCOMP(7,  m1.w, f1.w)
    WCOMP(8,  m2.x, f2.x) WCOMP(9,  m2.y, f2.y) WCOMP(10, m2.z, f2.z) WCOMP(11, m2.w, f2.w)
    WCOMP(12, m3.x, f3.x) WCOMP(13, m3.y, f3.y) WCOMP(14, m3.z, f3.z) WCOMP(15, m3.w, f3.w)
#undef WCOMP
    return A;
}

__global__ __launch_bounds__(128) void gat_attn(const _Float16* __restrict__ pT,
                                                const int* __restrict__ adj,
                                                const float* __restrict__ e,
                                                const float* __restrict__ f,
                                                const float* __restrict__ Mb,
                                                float* __restrict__ out)
{
    // Double-buffered panel: [buf][d=0..127][j_rel=0..31] f16  (2 x 8KB)
    __shared__ _Float16 ldsB[2][D_ * 32];

    const int t     = threadIdx.x;              // 0..127
    const int lane  = t & 31;
    const int wave  = t >> 5;
    const int tile2 = blockIdx.x * 4 + wave;    // 0..511, 32 rows each; WG = same b
    const int b     = tile2 >> 6;               // / (V/32)
    const int i0    = (tile2 & 63) << 5;
    const int row   = lane & 15;
    const int hi    = lane >> 4;

    const float mb  = Mb[b];
    float ei[2], mi[2];
#pragma unroll
    for (int k = 0; k < 2; ++k) {
        ei[k] = e[b * V_ + i0 + 16 * k + row];
        mi[k] = lrelu(ei[k] + mb);
    }

    const v8f zero = {};
    v8f acc[2][8];
#pragma unroll
    for (int k = 0; k < 2; ++k)
#pragma unroll
        for (int i = 0; i < 8; ++i) acc[k][i] = zero;
    float lsum[2] = {0.0f, 0.0f};

    const int*      adjr0 = adj + (size_t)(i0 + row) * V_;
    const int*      adjr1 = adjr0 + (size_t)16 * V_;
    const float*    fb    = f + b * V_;
    const _Float16* ptb   = pT + (size_t)b * D_ * V_;

    // Raw LDS byte offset of ldsB.
    const unsigned lds_base = (unsigned)(unsigned long long)(uintptr_t)&ldsB[0][0];
    // This thread DMAs row d = t of each panel: 64B = 4 x 16B.
    const unsigned my_lds0  = lds_base + (unsigned)t * 64u;

    // Prefetch chunk 0 into buffer 0.
    async_panel_16B_x4(my_lds0, (const void*)(ptb + (size_t)t * V_));

    for (int n = 0; n < NCHUNK; ++n) {
        const int jt  = n * 32;
        const int cur = n & 1;

        if (n + 1 < NCHUNK) {
            // Kick chunk n+1 into the other buffer, then wait for chunk n only.
            async_panel_16B_x4(my_lds0 + (unsigned)(!cur) * (D_ * 32 * 2),
                               (const void*)(ptb + (size_t)t * V_ + (jt + 32)));
            asm volatile("s_wait_asynccnt 0x4" ::: "memory");
        } else {
            asm volatile("s_wait_asynccnt 0x0" ::: "memory");
        }
        __syncthreads();   // chunk n visible to all waves

        // --- probability tiles (A, 16x32), f_j loads shared by both i-tiles ---
        const int j0 = jt + 16 * hi;        // A-layout K runs: j0..j0+15 contiguous
        const int j1 = j0 + 8;
        float4 f0 = *(const float4*)(fb + j0);
        float4 f1 = *(const float4*)(fb + j0 + 4);
        float4 f2 = *(const float4*)(fb + j1);
        float4 f3 = *(const float4*)(fb + j1 + 4);

        int4 a00 = *(const int4*)(adjr0 + j0);
        int4 a01 = *(const int4*)(adjr0 + j0 + 4);
        int4 a02 = *(const int4*)(adjr0 + j1);
        int4 a03 = *(const int4*)(adjr0 + j1 + 4);
        int4 a10 = *(const int4*)(adjr1 + j0);
        int4 a11 = *(const int4*)(adjr1 + j0 + 4);
        int4 a12 = *(const int4*)(adjr1 + j1);
        int4 a13 = *(const int4*)(adjr1 + j1 + 4);

        v16h A0 = build_A(ei[0], mi[0], lsum[0], f0, f1, f2, f3, a00, a01, a02, a03);
        v16h A1 = build_A(ei[1], mi[1], lsum[1], f0, f1, f2, f3, a10, a11, a12, a13);

        // --- 16 WMMAs against the LDS panel; each B tile feeds 2 WMMAs ---
#pragma unroll
        for (int dt = 0; dt < 8; ++dt) {
            // B tile (32x16): lane column d = dt*16+row, K = 16*hi + 0..15
            v16h Bm = *(const v16h*)&ldsB[cur][(dt * 16 + row) * 32 + hi * 16];
            acc[0][dt] = __builtin_amdgcn_wmma_f32_16x16x32_f16(
                false, A0, false, Bm, (short)0, acc[0][dt], false, false);
            acc[1][dt] = __builtin_amdgcn_wmma_f32_16x16x32_f16(
                false, A1, false, Bm, (short)0, acc[1][dt], false, false);
        }
        __syncthreads();   // all waves done reading buf[cur]; reusable for n+2
    }

    // Epilogue per 16-row tile: row sums, normalize, relu, store.
#pragma unroll
    for (int k = 0; k < 2; ++k) {
        // Lanes L and L+16 cover disjoint j halves of row (L&15).
        float ls = lsum[k] + __shfl_xor(lsum[k], 16, 32);
        // C layout rows: i0 + 16k + 8*hi + r. Row sum lives in lane (8*hi + r).
        float linv[8];
#pragma unroll
        for (int r = 0; r < 8; ++r) {
            float lr = __shfl(ls, 8 * hi + r, 32);
            linv[r] = 1.0f / lr;
        }
#pragma unroll
        for (int dt = 0; dt < 8; ++dt) {
            const int d = dt * 16 + row;
#pragma unroll
            for (int r = 0; r < 8; ++r) {
                const int i = i0 + 16 * k + 8 * hi + r;
                out[((size_t)b * V_ + i) * D_ + d] = fmaxf(acc[k][dt][r] * linv[r], 0.0f);
            }
        }
    }
}

// ---------------------------------------------------------------------------
extern "C" void kernel_launch(void* const* d_in, const int* in_sizes, int n_in,
                              void* d_out, int out_size, void* d_ws, size_t ws_size,
                              hipStream_t stream)
{
    const float* x   = (const float*)d_in[0];
    const int*   adj = (const int*)d_in[1];
    const float* W   = (const float*)d_in[2];
    const float* a   = (const float*)d_in[3];
    float*       out = (float*)d_out;

    char* ws = (char*)d_ws;
    _Float16* pT = (_Float16*)ws;                              // 4 MB
    float*    e  = (float*)(ws + (size_t)B_ * D_ * V_ * 2);    // 64 KB
    float*    f  = e + B_ * V_;                                // 64 KB
    float*    Mb = f + B_ * V_;                                // 32 B (+pad)
    _Float16* Wh = (_Float16*)(Mb + 64);                       // 64 KB, 256B-aligned

    gat_cvtw<<<(D_ * H_) / 256, 256, 0, stream>>>(W, Wh);
    gat_proj<<<128, 128, 0, stream>>>(x, Wh, pT);
    gat_ef  <<<(B_ * V_) / 256, 256, 0, stream>>>(pT, a, e, f);
    gat_max <<<B_, 256, 0, stream>>>(f, Mb);
    gat_attn<<<128, 128, 0, stream>>>(pT, adj, e, f, Mb, out);
}
